// TraderGRU_9629316678272
// MI455X (gfx1250) — compile-verified, matrix operand order checked by reading, not run on hardware
//
#include <hip/hip_runtime.h>
#include <hip/hip_bf16.h>
#include <math.h>

// ---------------------------------------------------------------------------
// TraderGRU on gfx1250: persistent per-batch-slice GRU scan using
// v_wmma_f32_16x16x32_bf16. 32 workgroups x 256 threads (8 wave32s).
// Each workgroup owns 16 batch rows for all 1024 timesteps; h lives in LDS,
// w_ih lives in LDS (96 KB), w_hh streams from L2 (384 KB bf16, L2-resident).
// Per K-tile, all 6 N-tile B-fragments are loaded into distinct registers
// (one clause + one wait + 6 back-to-back WMMAs), and the gh pass
// double-buffers fragments across K-tiles to hide L2/LDS latency.
// ---------------------------------------------------------------------------

typedef __attribute__((ext_vector_type(16))) __bf16 bf16x16;
typedef __attribute__((ext_vector_type(8)))  __bf16 bf16x8;
typedef __attribute__((ext_vector_type(8)))  float  f32x8;

union BFrag {
    bf16x16 v;
    bf16x8  h2[2];
    __bf16  e[16];
};

__device__ __forceinline__ void ld_frag(BFrag& f, const __bf16* p) {
    f.h2[0] = *reinterpret_cast<const bf16x8*>(p);
    f.h2[1] = *reinterpret_cast<const bf16x8*>(p + 16);
}

__device__ __forceinline__ float sigmoidf(float x) {
    return 1.0f / (1.0f + __expf(-x));
}

// Convert w_hh (fp32, [768,256]) to bf16 scratch once (hardware cvt).
__global__ void cvt_bf16_kernel(const float* __restrict__ in,
                                __bf16* __restrict__ out, int n) {
    int i = blockIdx.x * 256 + threadIdx.x;
    if (i < n) out[i] = (__bf16)in[i];
}

#define BB 512
#define TT 1024
#define FF 64
#define HH 256
#define GG 768   // 3*H

__global__ __launch_bounds__(256)
void tradergru_scan_kernel(const float* __restrict__ x,        // [B,T,F]
                           const float* __restrict__ noise,    // [T,B,1]
                           const float* __restrict__ w_ih,     // [3H,F]
                           const float* __restrict__ b_ih,     // [3H]
                           const float* __restrict__ b_hh,     // [3H]
                           const float* __restrict__ w_action, // [1,H]
                           const float* __restrict__ b_action, // [1]
                           const float* __restrict__ w_out,    // [1,H]
                           const float* __restrict__ b_out,    // [1]
                           const __bf16* __restrict__ whh_bf,  // [3H,H] bf16
                           float* __restrict__ out)            // [B,T]
{
    __shared__ float  lds_gx[16 * GG];   // 48 KB: x-side gate pre-acts (+b_ih)
    __shared__ float  lds_gh[16 * GG];   // 48 KB: h-side gate pre-acts (+b_hh)
    __shared__ float  h_f32[16 * HH];    // 16 KB: fp32 hidden state
    __shared__ __bf16 h_bf [16 * HH];    //  8 KB: bf16 hidden state (WMMA A)
    __shared__ __bf16 s_wih[GG * FF];    // 96 KB: w_ih bf16 (WMMA B, resident)
    __shared__ float  s_wact[HH];
    __shared__ float  s_wout[HH];

    const int tid  = threadIdx.x;
    const int lane = tid & 31;
    const int w    = tid >> 5;       // wave id 0..7
    const int half = lane >> 4;      // 0: lanes 0-15, 1: lanes 16-31
    const int nm   = lane & 15;      // row (A) / col (B) index within tile
    const int b0   = blockIdx.x * 16;

    // ---- prologue: stage vectors/weights into LDS, zero h ----
    s_wact[tid] = w_action[tid];
    s_wout[tid] = w_out[tid];
    for (int e = tid; e < 16 * HH; e += 256) {
        h_f32[e] = 0.0f;
        h_bf[e]  = (__bf16)0.0f;
    }
    for (int e = tid; e < GG * FF; e += 256)   // 192 elems/thread, hw cvt
        s_wih[e] = (__bf16)w_ih[e];

    float bih[6], bhh[6];
    for (int i = 0; i < 6; ++i) {
        const int g = (w * 6 + i) * 16 + nm;   // gate row this lane supplies
        bih[i] = b_ih[g];
        bhh[i] = b_hh[g];
    }
    __syncthreads();

    // lane-invariant fragment base pointers (chunk = base + kt*32 + {0,16})
    const float*  xrow  = x + (size_t)(b0 + nm) * (TT * FF) + half * 8;
    const __bf16* hbase = h_bf + nm * HH + half * 8;
    const __bf16* wbase = whh_bf + (size_t)(w * 96 + nm) * HH + half * 8;
    const __bf16* ibase = s_wih + (w * 96 + nm) * FF + half * 8;
    const float   bact  = b_action[0];
    const float   bout  = b_out[0];

    for (int t = 0; t < TT; ++t) {
        // ============ phase 1a: gx = x_t @ w_ih^T (2 K-tiles) ============
        {
            f32x8 acc[6];
            for (int i = 0; i < 6; ++i)
                for (int r = 0; r < 8; ++r) acc[i][r] = bih[i];

            const float* xp = xrow + t * FF;
            for (int kt = 0; kt < 2; ++kt) {
                BFrag a;
                const float* p = xp + kt * 32;
                for (int q = 0; q < 8; ++q) {     // hw v_cvt_pk_bf16_f32
                    a.e[q]     = (__bf16)p[q];
                    a.e[8 + q] = (__bf16)p[16 + q];
                }
                BFrag bw[6];                      // distinct regs -> one clause
                for (int i = 0; i < 6; ++i)
                    ld_frag(bw[i], ibase + i * 16 * FF + kt * 32);
                for (int i = 0; i < 6; ++i)
                    acc[i] = __builtin_amdgcn_wmma_f32_16x16x32_bf16(
                        false, a.v, false, bw[i].v, (short)0, acc[i], false, false);
            }
            for (int i = 0; i < 6; ++i) {
                const int col = (w * 6 + i) * 16 + nm;
                for (int r = 0; r < 8; ++r)
                    lds_gx[(r + half * 8) * GG + col] = acc[i][r];
            }
        }

        if (t + 1 < TT) __builtin_prefetch(xrow + (t + 1) * FF, 0, 1);

        // ============ phase 1b: gh = h @ w_hh^T (8 K-tiles) ============
        // Double-buffered: kt+1 fragment loads issue before the kt WMMA
        // chain so L2/LDS latency overlaps matrix math.
        {
            f32x8 acc[6];
            for (int i = 0; i < 6; ++i)
                for (int r = 0; r < 8; ++r) acc[i][r] = bhh[i];

            BFrag aP[2], bP[2][6];
            ld_frag(aP[0], hbase);
            for (int i = 0; i < 6; ++i)
                ld_frag(bP[0][i], wbase + i * 16 * HH);

            #pragma unroll
            for (int kt = 0; kt < 8; ++kt) {
                const int cur = kt & 1, nxt = cur ^ 1;
                if (kt < 7) {
                    ld_frag(aP[nxt], hbase + (kt + 1) * 32);
                    for (int i = 0; i < 6; ++i)
                        ld_frag(bP[nxt][i], wbase + i * 16 * HH + (kt + 1) * 32);
                }
                for (int i = 0; i < 6; ++i)
                    acc[i] = __builtin_amdgcn_wmma_f32_16x16x32_bf16(
                        false, aP[cur].v, false, bP[cur][i].v, (short)0,
                        acc[i], false, false);
            }
            for (int i = 0; i < 6; ++i) {
                const int col = (w * 6 + i) * 16 + nm;
                for (int r = 0; r < 8; ++r)
                    lds_gh[(r + half * 8) * GG + col] = acc[i][r];
            }
        }
        __syncthreads();

        // ================= phase 2: gates / h update =================
        for (int e = tid; e < 16 * HH; e += 256) {
            const int b = e >> 8;        // local batch row 0..15
            const int k = e & 255;       // hidden index
            const float rg = sigmoidf(lds_gx[b * GG + k]       + lds_gh[b * GG + k]);
            const float zg = sigmoidf(lds_gx[b * GG + 256 + k] + lds_gh[b * GG + 256 + k]);
            const float ng = tanhf(lds_gx[b * GG + 512 + k] + rg * lds_gh[b * GG + 512 + k]);
            const float ho = h_f32[b * HH + k];
            const float hn = (1.0f - zg) * ng + zg * ho;
            h_f32[b * HH + k] = hn;
            h_bf[b * HH + k]  = (__bf16)hn;
        }
        __syncthreads();

        // ================= phase 3: per-row outputs =================
        // 8 waves x 2 rows: wave32 tree reduction of two H=256 dot products.
        for (int rr = 0; rr < 2; ++rr) {
            const int brow = w * 2 + rr;
            float sa = 0.0f, so = 0.0f;
            for (int j = 0; j < 8; ++j) {
                const float hv = h_f32[brow * HH + lane + 32 * j];
                sa += hv * s_wact[lane + 32 * j];
                so += hv * s_wout[lane + 32 * j];
            }
            for (int m = 16; m >= 1; m >>= 1) {
                sa += __shfl_xor(sa, m, 32);
                so += __shfl_xor(so, m, 32);
            }
            if (lane == 0) {
                const float logit = sa + bact;
                const float nz    = noise[t * BB + (b0 + brow)];
                const float y = sigmoidf((logit + __logf(nz + 1e-20f)
                                                - __logf(1.0f - nz + 1e-20f)) * 20.0f);
                out[(size_t)(b0 + brow) * TT + t] = tanhf(so + bout) * y;
            }
        }
        // next phase-1 only writes lds_gx/gh (not read here); top-of-loop work
        // is fenced by this iteration's barriers — no extra barrier needed.
    }
}

extern "C" void kernel_launch(void* const* d_in, const int* in_sizes, int n_in,
                              void* d_out, int out_size, void* d_ws, size_t ws_size,
                              hipStream_t stream) {
    const float* x        = (const float*)d_in[0];
    const float* noise    = (const float*)d_in[1];
    const float* w_ih     = (const float*)d_in[2];
    const float* w_hh     = (const float*)d_in[3];
    const float* b_ih     = (const float*)d_in[4];
    const float* b_hh     = (const float*)d_in[5];
    const float* w_action = (const float*)d_in[6];
    const float* b_action = (const float*)d_in[7];
    const float* w_out    = (const float*)d_in[8];
    const float* b_out    = (const float*)d_in[9];
    float* outp = (float*)d_out;

    __bf16* whh_bf = (__bf16*)d_ws;            // 768*256*2 = 384 KB scratch
    const int n = GG * HH;
    cvt_bf16_kernel<<<(n + 255) / 256, 256, 0, stream>>>(w_hh, whh_bf, n);

    tradergru_scan_kernel<<<BB / 16, 256, 0, stream>>>(
        x, noise, w_ih, b_ih, b_hh, w_action, b_action, w_out, b_out,
        whh_bf, outp);
}
